// Attention_24111946400343
// MI455X (gfx1250) — compile-verified, matrix operand order checked by reading, not run on hardware
//
#include <hip/hip_runtime.h>
#include <hip/hip_bf16.h>
#include <math.h>

// ---------------------------------------------------------------------------
// Multi-head attention forward, CDNA5 (gfx1250), wave32 + WMMA f16->f32.
//   B=8, S=1024, D=768, H=12, HD=64, scale = 1/8.
// Pipeline: cvt(x,W) -> QKV WMMA GEMM -> flash attention (WMMA + async-LDS)
//           -> proj WMMA GEMM
// ---------------------------------------------------------------------------

#define BQ   8
#define SQ   1024
#define DQ   768
#define HQ   12
#define HDQ  64
#define NQKV 2304
#define MTOT (BQ * SQ)           // 8192
#define ATT_SCALE 0.125f

typedef __attribute__((ext_vector_type(16))) _Float16 v16h;
typedef __attribute__((ext_vector_type(8)))  _Float16 v8h;
typedef __attribute__((ext_vector_type(8)))  float    v8f;

// Workspace layout (in _Float16 elements)
#define OFF_XH     0u                    // 8192*768
#define OFF_WQKV   6291456u              // 2304*768  (transposed: [n][k])
#define OFF_WPROJ  8060928u              // 768*768   (transposed: [n][k])
#define OFF_Q      8650752u              // (B*H, S, HD)
#define OFF_K      14942208u             // (B*H, S, HD)
#define OFF_VT     21233664u             // (B*H, HD, S)   -- V transposed
#define OFF_CTX    27525120u             // (B*S, D) merged heads

__device__ __forceinline__ v16h cat8(v8h lo, v8h hi) {
  return __builtin_shufflevector(lo, hi, 0, 1, 2, 3, 4, 5, 6, 7,
                                         8, 9, 10, 11, 12, 13, 14, 15);
}

__device__ __forceinline__ v16h ldA32(const _Float16* base, int half) {
  // 16-bit A/B fragment for a K-chunk of 32: element e -> K = e + 8*half (e<8)
  // or 16 + (e-8) + 8*half (e>=8). Two 16-byte contiguous loads per lane.
  v8h lo = *(const v8h*)(base + 8 * half);
  v8h hi = *(const v8h*)(base + 16 + 8 * half);
  return cat8(lo, hi);
}

#define WMMA_F16(a, b, c) \
  __builtin_amdgcn_wmma_f32_16x16x32_f16(false, (a), false, (b), (short)0, (c), false, false)

// gfx1250 async DMA: global -> LDS, 16 bytes per lane, tracked by ASYNCcnt.
__device__ __forceinline__ void async16(unsigned int ldsAddr, unsigned int goff,
                                        const _Float16* base) {
  asm volatile("global_load_async_to_lds_b128 %0, %1, %2"
               :
               : "v"(ldsAddr), "v"(goff), "s"(base)
               : "memory");
}

__device__ __forceinline__ void wait_async0() {
  asm volatile("s_wait_asynccnt 0" ::: "memory");
}

// ---------------------------------------------------------------------------
// Conversions
// ---------------------------------------------------------------------------
__global__ void cvt_f32_to_f16(const float* __restrict__ in,
                               _Float16* __restrict__ out, int n) {
  int i = blockIdx.x * blockDim.x + threadIdx.x;
  if (i < n) out[i] = (_Float16)in[i];
}

// in: [rows][cols] f32 ; out: [cols][rows] f16
__global__ void cvt_transpose_f16(const float* __restrict__ in,
                                  _Float16* __restrict__ out,
                                  int rows, int cols) {
  int i = blockIdx.x * blockDim.x + threadIdx.x;
  if (i < rows * cols) {
    int n = i / rows;     // output row  (0..cols)
    int k = i % rows;     // output col  (0..rows)
    out[i] = (_Float16)in[k * cols + n];
  }
}

// ---------------------------------------------------------------------------
// QKV GEMM: (8192 x 768) x (768 x 2304) + bias, scattered into Q/K/Vt (f16).
// One wave computes a 32(M) x 64(N) strip: 2 M-subtiles x 4 N-tiles = 8 acc,
// so every B fragment feeds two WMMAs. Fully unrolled k-loop lets the
// compiler hoist/clause the B128 loads ahead of the WMMAs.
// block = 4 waves. grid = (256, 9).
// ---------------------------------------------------------------------------
__global__ void __launch_bounds__(128)
qkv_gemm(const _Float16* __restrict__ xh,   // [8192][768]
         const _Float16* __restrict__ wT,   // [2304][768]
         const float* __restrict__ bias,    // [2304]
         _Float16* __restrict__ qb,         // (B*H,S,HD)
         _Float16* __restrict__ kb,         // (B*H,S,HD)
         _Float16* __restrict__ vt)         // (B*H,HD,S)
{
  const int lane = threadIdx.x & 31;
  const int wid  = threadIdx.x >> 5;
  const int m16  = lane & 15;
  const int half = lane >> 4;
  const int tileM = blockIdx.x;                       // 0..255 (32 rows each)
  const int n0    = (blockIdx.y * 4 + wid) * 64;      // N-strip base

  v8f acc[2][4] = {};
  const _Float16* a0p = xh + (size_t)(tileM * 32 + m16) * DQ;
  const _Float16* a1p = a0p + (size_t)16 * DQ;
  const _Float16* bp[4];
#pragma unroll
  for (int t = 0; t < 4; ++t) bp[t] = wT + (size_t)(n0 + t * 16 + m16) * DQ;

#pragma unroll
  for (int k0 = 0; k0 < DQ; k0 += 32) {
    if (k0 + 64 < DQ) __builtin_prefetch(a0p + k0 + 64, 0, 3);
    const v16h a0 = ldA32(a0p + k0, half);
    const v16h a1 = ldA32(a1p + k0, half);
#pragma unroll
    for (int t = 0; t < 4; ++t) {
      const v16h b = ldA32(bp[t] + k0, half);
      acc[0][t] = WMMA_F16(a0, b, acc[0][t]);
      acc[1][t] = WMMA_F16(a1, b, acc[1][t]);
    }
  }

  const int which = n0 / DQ;            // 0=Q 1=K 2=V (strip never crosses)
  const int h     = (n0 % DQ) / HDQ;    // head          (strip never crosses)
#pragma unroll
  for (int t = 0; t < 4; ++t) {
    const int n  = n0 + t * 16 + m16;
    const int hd = t * 16 + m16;        // (n % 768) % 64 for 64-aligned strips
    const float bn = bias[n];
#pragma unroll
    for (int sub = 0; sub < 2; ++sub) {
#pragma unroll
      for (int r = 0; r < 8; ++r) {
        const int m  = tileM * 32 + sub * 16 + r + 8 * half;
        const int bb = m >> 10;
        const int s  = m & 1023;
        const _Float16 hv = (_Float16)(acc[sub][t][r] + bn);
        if (which == 0)
          qb[((size_t)(bb * HQ + h) * SQ + s) * HDQ + hd] = hv;
        else if (which == 1)
          kb[((size_t)(bb * HQ + h) * SQ + s) * HDQ + hd] = hv;
        else
          vt[((size_t)(bb * HQ + h) * HDQ + hd) * SQ + s] = hv;
      }
    }
  }
}

// ---------------------------------------------------------------------------
// Flash attention. block = 4 waves, each wave owns 16 query rows of one (b,h).
// K/V key-chunks (32 keys) are staged cooperatively into LDS with
// global_load_async_to_lds_b128, double-buffered across the key loop:
// all 4 waves consume the same chunk.  grid = (16, 96).
// ---------------------------------------------------------------------------

// Stage one key-chunk: K tile 32x64 halves (contiguous 4 KB) and V tile
// 64 rows x 32 halves from Vt (row stride S). 128 threads x 4 async b128 ops.
__device__ __forceinline__ void stage_kv(const _Float16* Kbh, const _Float16* Vbh,
                                         int jj, _Float16* kDst, _Float16* vDst,
                                         int tid) {
  const unsigned int kLds = (unsigned int)(unsigned long long)kDst;
  const unsigned int vLds = (unsigned int)(unsigned long long)vDst;
  const _Float16* kBase = Kbh + (size_t)jj * HDQ;   // contiguous 4096 B
  const _Float16* vBase = Vbh + jj;
  const unsigned int o0 = (unsigned int)tid * 16u;  // bytes
  async16(kLds + o0, o0, kBase);
  async16(kLds + o0 + 2048u, o0 + 2048u, kBase);
#pragma unroll
  for (int i = 0; i < 2; ++i) {
    const int c = tid * 2 + i;                      // 0..255
    const unsigned int row  = (unsigned int)(c >> 2);
    const unsigned int part = (unsigned int)(c & 3);
    const unsigned int goff = (row * SQ + part * 8u) * 2u;   // bytes
    const unsigned int loff = (row * 32u + part * 8u) * 2u;  // bytes
    async16(vLds + loff, goff, vBase);
  }
}

__global__ void __launch_bounds__(128)
flash_attn(const _Float16* __restrict__ q,   // (B*H,S,HD)
           const _Float16* __restrict__ k,   // (B*H,S,HD)
           const _Float16* __restrict__ vt,  // (B*H,HD,S)
           _Float16* __restrict__ ctx)       // (B*S,D)
{
  const int tid  = threadIdx.x;
  const int lane = tid & 31;
  const int wid  = tid >> 5;
  const int m16  = lane & 15;
  const int half = lane >> 4;
  const int qblk = blockIdx.x * 4 + wid;     // 0..63
  const int bh   = blockIdx.y;               // 0..95

  const _Float16* Q = q  + (size_t)bh * SQ * HDQ;
  const _Float16* K = k  + (size_t)bh * SQ * HDQ;
  const _Float16* V = vt + (size_t)bh * HDQ * SQ;

  __shared__ __align__(16) _Float16 kbuf[2][32 * HDQ];   // 2 x 4 KB
  __shared__ __align__(16) _Float16 vbuf[2][HDQ * 32];   // 2 x 4 KB
  __shared__ __align__(16) _Float16 pbuf[4][16 * 32];    // per-wave P staging
  _Float16* pb = pbuf[wid];

  // Q fragments (16 x 64, split into two K=32 A-fragments)
  const _Float16* qrow = Q + (size_t)(qblk * 16 + m16) * HDQ;
  const v16h qa0 = ldA32(qrow, half);
  const v16h qa1 = ldA32(qrow + 32, half);

  v8f o[4] = {};
  float rowmax[8], rowsum[8];
#pragma unroll
  for (int r = 0; r < 8; ++r) { rowmax[r] = -INFINITY; rowsum[r] = 0.0f; }

  stage_kv(K, V, 0, kbuf[0], vbuf[0], tid);

  for (int jj = 0; jj < SQ; jj += 32) {
    const int cur = (jj >> 5) & 1;
    wait_async0();           // our async ops for kbuf[cur]/vbuf[cur] complete
    __syncthreads();         // every wave's portion complete + prev reads done
    if (jj + 32 < SQ)        // overlap next chunk's DMA with this chunk's math
      stage_kv(K, V, jj + 32, kbuf[cur ^ 1], vbuf[cur ^ 1], tid);

    const _Float16* kT = kbuf[cur];   // [32][64]
    const _Float16* vT = vbuf[cur];   // [64][32]

    // ---- scores: S = Q * K^T  (two 16x16 tiles, K-dim = HD = 64) ----
    const _Float16* kr0 = kT + m16 * HDQ;
    const _Float16* kr1 = kT + (16 + m16) * HDQ;
    v8f s0 = {}, s1 = {};
    s0 = WMMA_F16(qa0, ldA32(kr0, half),      s0);
    s0 = WMMA_F16(qa1, ldA32(kr0 + 32, half), s0);
    s1 = WMMA_F16(qa0, ldA32(kr1, half),      s1);
    s1 = WMMA_F16(qa1, ldA32(kr1 + 32, half), s1);

    // ---- online softmax over the 32 new columns ----
    // C-layout: lane holds row M = r + 8*half, col = m16 (+16 for tile 1).
    // Row reduction = shfl-xor over the 16 lanes of this half (masks 1..8).
#pragma unroll
    for (int r = 0; r < 8; ++r) {
      float a = s0[r] * ATT_SCALE;
      float b = s1[r] * ATT_SCALE;
      float mx = fmaxf(a, b);
#pragma unroll
      for (int mask = 1; mask < 16; mask <<= 1)
        mx = fmaxf(mx, __shfl_xor(mx, mask, 32));
      const float mnew  = fmaxf(rowmax[r], mx);
      const float alpha = __expf(rowmax[r] - mnew);
      const float p0 = __expf(a - mnew);
      const float p1 = __expf(b - mnew);
      float ps = p0 + p1;
#pragma unroll
      for (int mask = 1; mask < 16; mask <<= 1)
        ps += __shfl_xor(ps, mask, 32);
      rowsum[r] = rowsum[r] * alpha + ps;
      rowmax[r] = mnew;
#pragma unroll
      for (int t = 0; t < 4; ++t) o[t][r] *= alpha;
      // stage P (16x32 f16) in LDS, row-major, for C->A relayout (intra-wave)
      pb[(r + 8 * half) * 32 + m16]      = (_Float16)p0;
      pb[(r + 8 * half) * 32 + 16 + m16] = (_Float16)p1;
    }

    // ---- O += P(16x32) * Vchunk(32x64) ----
    const v16h pa = ldA32(pb + m16 * 32, half);   // A-layout read of P
#pragma unroll
    for (int t = 0; t < 4; ++t) {
      const _Float16* vr = vT + (t * 16 + m16) * 32;
      o[t] = WMMA_F16(pa, ldA32(vr, half), o[t]);
    }
  }

  // ---- normalize and store context (merged heads, f16) ----
  const int bb = bh / HQ;
  const int h  = bh % HQ;
#pragma unroll
  for (int r = 0; r < 8; ++r) {
    const float inv = 1.0f / rowsum[r];
    const int s = qblk * 16 + r + 8 * half;
#pragma unroll
    for (int t = 0; t < 4; ++t)
      ctx[(size_t)(bb * SQ + s) * DQ + h * HDQ + t * 16 + m16] =
          (_Float16)(o[t][r] * inv);
  }
}

// ---------------------------------------------------------------------------
// Output projection: (8192 x 768) x (768 x 768) + bias -> fp32 d_out.
// Same 32x64-per-wave scheme. grid = (256, 3), block = 128.
// ---------------------------------------------------------------------------
__global__ void __launch_bounds__(128)
proj_gemm(const _Float16* __restrict__ ctx,  // [8192][768]
          const _Float16* __restrict__ wT,   // [768][768] (transposed)
          const float* __restrict__ bias,    // [768]
          float* __restrict__ out)           // [8192][768]
{
  const int lane = threadIdx.x & 31;
  const int wid  = threadIdx.x >> 5;
  const int m16  = lane & 15;
  const int half = lane >> 4;
  const int tileM = blockIdx.x;
  const int n0    = (blockIdx.y * 4 + wid) * 64;

  v8f acc[2][4] = {};
  const _Float16* a0p = ctx + (size_t)(tileM * 32 + m16) * DQ;
  const _Float16* a1p = a0p + (size_t)16 * DQ;
  const _Float16* bp[4];
#pragma unroll
  for (int t = 0; t < 4; ++t) bp[t] = wT + (size_t)(n0 + t * 16 + m16) * DQ;

#pragma unroll
  for (int k0 = 0; k0 < DQ; k0 += 32) {
    if (k0 + 64 < DQ) __builtin_prefetch(a0p + k0 + 64, 0, 3);
    const v16h a0 = ldA32(a0p + k0, half);
    const v16h a1 = ldA32(a1p + k0, half);
#pragma unroll
    for (int t = 0; t < 4; ++t) {
      const v16h b = ldA32(bp[t] + k0, half);
      acc[0][t] = WMMA_F16(a0, b, acc[0][t]);
      acc[1][t] = WMMA_F16(a1, b, acc[1][t]);
    }
  }

#pragma unroll
  for (int t = 0; t < 4; ++t) {
    const int n = n0 + t * 16 + m16;
    const float bn = bias[n];
#pragma unroll
    for (int sub = 0; sub < 2; ++sub) {
#pragma unroll
      for (int r = 0; r < 8; ++r) {
        const int m = tileM * 32 + sub * 16 + r + 8 * half;
        out[(size_t)m * DQ + n] = acc[sub][t][r] + bn;
      }
    }
  }
}

// ---------------------------------------------------------------------------
extern "C" void kernel_launch(void* const* d_in, const int* in_sizes, int n_in,
                              void* d_out, int out_size, void* d_ws,
                              size_t ws_size, hipStream_t stream) {
  const float* x      = (const float*)d_in[0];   // (8,1024,768)
  const float* W_qkv  = (const float*)d_in[1];   // (768,2304)
  const float* b_qkv  = (const float*)d_in[2];   // (2304)
  const float* W_proj = (const float*)d_in[3];   // (768,768)
  const float* b_proj = (const float*)d_in[4];   // (768)
  float* out = (float*)d_out;

  _Float16* ws     = (_Float16*)d_ws;
  _Float16* xh     = ws + OFF_XH;
  _Float16* wqkvT  = ws + OFF_WQKV;
  _Float16* wprojT = ws + OFF_WPROJ;
  _Float16* qb     = ws + OFF_Q;
  _Float16* kb     = ws + OFF_K;
  _Float16* vt     = ws + OFF_VT;
  _Float16* ctx    = ws + OFF_CTX;

  const int nX  = MTOT * DQ;         // 6291456
  const int nWq = DQ * NQKV;         // 1769472
  const int nWp = DQ * DQ;           // 589824

  cvt_f32_to_f16<<<(nX + 255) / 256, 256, 0, stream>>>(x, xh, nX);
  cvt_transpose_f16<<<(nWq + 255) / 256, 256, 0, stream>>>(W_qkv, wqkvT, DQ, NQKV);
  cvt_transpose_f16<<<(nWp + 255) / 256, 256, 0, stream>>>(W_proj, wprojT, DQ, DQ);

  qkv_gemm<<<dim3(MTOT / 32, NQKV / 256), 128, 0, stream>>>(
      xh, wqkvT, b_qkv, qb, kb, vt);

  flash_attn<<<dim3(SQ / 64, BQ * HQ), 128, 0, stream>>>(qb, kb, vt, ctx);

  proj_gemm<<<dim3(MTOT / 32, DQ / 256), 128, 0, stream>>>(
      ctx, wprojT, b_proj, out);
}
